// LMFEAM_SSN_59657095741471
// MI455X (gfx1250) — matrix operand order repaired and checked by reading, not compiled, not run on hardware
//
#include <hip/hip_runtime.h>
#include <hip/hip_bf16.h>
#include <math.h>

// ---------------------------------------------------------------------------
// CDNA5 / gfx1250 implementation of LMFEAM + soft-SLIC.
// All GEMM-shaped work: v_wmma_f32_16x16x32_f16 (wave32, 16x16 tiles), with
// per-WGP async staging of A-tiles into LDS (GLOBAL_LOAD_ASYNC_TO_LDS_B128).
// ---------------------------------------------------------------------------

typedef __attribute__((ext_vector_type(16))) _Float16 v16h;
typedef __attribute__((ext_vector_type(8)))  float    v8f;

__device__ __forceinline__ v8f wmma32(v16h a, v16h b, v8f c) {
  // (neg_a, A, neg_b, B, c_mod, C, reuse_a, reuse_b)
  return __builtin_amdgcn_wmma_f32_16x16x32_f16(false, a, false, b, (short)0, c,
                                                false, false);
}

// 16-bit A-matrix 16x32 per-lane K mapping (contiguous, matches B loads):
//   lanes 0-15 : elements 0..15 -> K 0..15
//   lanes16-31 : elements 0..15 -> K 16..31
__device__ __forceinline__ int kkmap(int j, bool hi) {
  return j + (hi ? 16 : 0);
}

#define NPTS 8192
#define KNN  32
#define NSP  256

// ---------------------------------------------------------------------------
// Async global -> LDS staging (gfx1250 ASYNCcnt path) with safe fallback.
// Builtin param types (from clang diagnostic): int4 AS(1)* src, int4 AS(3)* dst
// ---------------------------------------------------------------------------
#if defined(__HIP_DEVICE_COMPILE__) && \
    __has_builtin(__builtin_amdgcn_global_load_async_to_lds_b128)
#define HAVE_ASYNC_LDS 1
#else
#define HAVE_ASYNC_LDS 0
#endif

typedef int v4i_vs __attribute__((vector_size(16)));
typedef v4i_vs __attribute__((address_space(1)))* gv4ip;
typedef v4i_vs __attribute__((address_space(3)))* lv4ip;

__device__ __forceinline__ void stage_lds(const void* gsrc, void* ldst,
                                          int bytes, int tid, int nthreads) {
#if HAVE_ASYNC_LDS
  for (int off = tid * 16; off < bytes; off += nthreads * 16) {
    __builtin_amdgcn_global_load_async_to_lds_b128(
        (gv4ip)((const char*)gsrc + off),
        (lv4ip)((char*)ldst + off), 0, 0);
  }
#if __has_builtin(__builtin_amdgcn_s_wait_asynccnt)
  __builtin_amdgcn_s_wait_asynccnt(0);
#else
  asm volatile("s_wait_asynccnt 0" ::: "memory");
#endif
#else
  const uint4* s = (const uint4*)gsrc;
  uint4* d = (uint4*)ldst;
  for (int i = tid; i < bytes / 16; i += nthreads) d[i] = s[i];
#endif
}

// ---------------------------------------------------------------------------
// Weight packer: fp32 row-major W[M][K] -> per-lane WMMA A-layout f16 tiles.
// Flat index: ((mt*nKt + kt)*32 + lane)*16 + j
// ---------------------------------------------------------------------------
__global__ void k_pack_w(const float* __restrict__ W, _Float16* __restrict__ Wp,
                         int M, int K, int Mpad, int Kpad) {
  int t = blockIdx.x * blockDim.x + threadIdx.x;
  int nKt = Kpad >> 5;
  int total = (Mpad >> 4) * nKt * 512;
  if (t >= total) return;
  int j    = t & 15;
  int lane = (t >> 4) & 31;
  int tile = t >> 9;
  int kt = tile % nKt;
  int mt = tile / nKt;
  int m  = (mt << 4) + (lane & 15);
  int k  = (kt << 5) + kkmap(j, lane >= 16);
  float v = (m < M && k < K) ? W[(size_t)m * K + k] : 0.f;
  Wp[t] = (_Float16)v;
}

// Pack transposed c (CBUF is [32 ch][256 sp]) into A-layout: A[m=s][k=c].
__global__ void k_pack_cT(const float* __restrict__ CBUF,
                          _Float16* __restrict__ Ap) {
  int t = blockIdx.x * blockDim.x + threadIdx.x;   // 16 tiles * 512 = 8192
  int j = t & 15, lane = (t >> 4) & 31, mt = t >> 9;
  int m = (mt << 4) + (lane & 15);                 // superpixel s
  int k = kkmap(j, lane >= 16);                    // channel c
  float v = (k < 20) ? CBUF[k * 256 + m] : 0.f;
  Ap[t] = (_Float16)v;
}

// ---------------------------------------------------------------------------
// f1 = relu(W1b @ relu(W1a @ x + b1a) + b1b), stored point-major f16 [N][64]
// ---------------------------------------------------------------------------
__global__ void k_f1(const float* __restrict__ x,
                     const float* __restrict__ w1a, const float* __restrict__ b1a,
                     const float* __restrict__ w1b, const float* __restrict__ b1b,
                     _Float16* __restrict__ F1H) {
  int n = blockIdx.x * blockDim.x + threadIdx.x;
  float x0 = x[n], x1 = x[NPTS + n], x2 = x[2 * NPTS + n];
  float h[32];
  for (int o = 0; o < 32; ++o)
    h[o] = fmaxf(b1a[o] + w1a[o * 3] * x0 + w1a[o * 3 + 1] * x1 +
                     w1a[o * 3 + 2] * x2, 0.f);
  for (int o = 0; o < 64; ++o) {
    float s = b1b[o];
    for (int c = 0; c < 32; ++c) s += w1b[o * 32 + c] * h[c];
    F1H[(size_t)n * 64 + o] = (_Float16)fmaxf(s, 0.f);
  }
}

// ---------------------------------------------------------------------------
// KNN: per-thread top-32 insertion over LDS-staged candidate tiles.
// ---------------------------------------------------------------------------
__global__ void k_knn(const float* __restrict__ x, float* __restrict__ D2N,
                      int* __restrict__ IDXb) {
  __shared__ float sx[256], sy[256], sz[256];
  int tid = threadIdx.x;
  int q = blockIdx.x * 256 + tid;
  float qx = x[q], qy = x[NPTS + q], qz = x[2 * NPTS + q];
  float bd[KNN];
  int   bi[KNN];
  for (int k = 0; k < KNN; ++k) { bd[k] = 1e30f; bi[k] = 0; }
  for (int tile = 0; tile < NPTS / 256; ++tile) {
    int c = tile * 256 + tid;
    sx[tid] = x[c]; sy[tid] = x[NPTS + c]; sz[tid] = x[2 * NPTS + c];
    __syncthreads();
    for (int j = 0; j < 256; ++j) {
      float dx = qx - sx[j], dy = qy - sy[j], dz = qz - sz[j];
      float d2 = dx * dx + dy * dy + dz * dz;
      if (d2 < bd[KNN - 1]) {
        int pos = KNN - 1;
        while (pos > 0 && bd[pos - 1] > d2) {
          bd[pos] = bd[pos - 1]; bi[pos] = bi[pos - 1]; --pos;
        }
        bd[pos] = d2; bi[pos] = tile * 256 + j;
      }
    }
    __syncthreads();
  }
  for (int k = 0; k < KNN; ++k) {
    D2N[(size_t)q * KNN + k] = bd[k];
    IDXb[(size_t)q * KNN + k] = bi[k];
  }
}

// ---------------------------------------------------------------------------
// Neighbor MLP: h2 = relu(wm2 @ relu(wm1 @ gather(f1) + bm1) + bm2)
// WMMA GEMM M=64,K=64 over 262144 gathered columns.
// wm1 A-tiles async-staged into LDS once per block; B reused across 4 mtiles.
// ---------------------------------------------------------------------------
__global__ void k_edge(const _Float16* __restrict__ Wp_wm1,
                       const float* __restrict__ bm1,
                       const float* __restrict__ wm2,
                       const float* __restrict__ bm2,
                       const _Float16* __restrict__ F1H,
                       const int* __restrict__ IDXb,
                       float* __restrict__ H2) {
  __shared__ __align__(16) _Float16 wlds[8 * 512];   // 8 A-tiles (4 mt x 2 kt)
  __shared__ float hbuf[4][64][16];
  stage_lds(Wp_wm1, wlds, 8 * 1024, threadIdx.x, 128);
  __syncthreads();
  int w = threadIdx.x >> 5, L = threadIdx.x & 31;
  int ct = blockIdx.x * 4 + w;                 // column tile (16 cols)
  int col = (ct << 4) + (L & 15);              // global column = n*32 + knb
  int p = IDXb[col];                           // gathered point index
  int kbase = (L < 16) ? 0 : 16;
  v8f acc[4] = {};
  for (int kt = 0; kt < 2; ++kt) {
    v16h b = *(const v16h*)(F1H + (size_t)p * 64 + kt * 32 + kbase);
#pragma unroll
    for (int mt = 0; mt < 4; ++mt) {
      v16h a = *(const v16h*)(wlds + ((mt * 2 + kt) * 32 + L) * 16);
      acc[mt] = wmma32(a, b, acc[mt]);
    }
  }
  int cl = L & 15;
  int rb = (L < 16) ? 0 : 8;
#pragma unroll
  for (int mt = 0; mt < 4; ++mt)
    for (int v = 0; v < 8; ++v) {
      int row = mt * 16 + rb + v;
      hbuf[w][row][cl] = fmaxf(acc[mt][v] + bm1[row], 0.f);
    }
  __syncthreads();
  for (int i = 0; i < 3; ++i) {               // 96 tasks / 32 lanes
    int t = i * 32 + L;
    int c2 = t & 15;                           // local column
    int r2 = t >> 4;                           // output row 0..5
    float s = bm2[r2];
    for (int c = 0; c < 64; ++c) s += wm2[r2 * 64 + c] * hbuf[w][c][c2];
    H2[(size_t)((ct << 4) + c2) * 8 + r2] = fmaxf(s, 0.f);
  }
}

// ---------------------------------------------------------------------------
// Multi-scale masked max over neighbors -> msf [18][N] (written to d_out)
// ---------------------------------------------------------------------------
__global__ void k_msf(const float* __restrict__ H2,
                      const float* __restrict__ D2N,
                      float* __restrict__ msf) {
  int n = blockIdx.x * blockDim.x + threadIdx.x;
  const float r2s[3] = {0.04f, 0.16f, 0.36f};
  for (int s = 0; s < 3; ++s) {
    float best[6] = {-1e30f, -1e30f, -1e30f, -1e30f, -1e30f, -1e30f};
    for (int k = 0; k < KNN; ++k) {
      if (D2N[(size_t)n * KNN + k] <= r2s[s]) {
        const float* hp = H2 + (size_t)(n * KNN + k) * 8;
        for (int j = 0; j < 6; ++j) best[j] = fmaxf(best[j], hp[j]);
      }
    }
    for (int j = 0; j < 6; ++j)
      msf[(size_t)(s * 6 + j) * NPTS + n] = best[j];
  }
}

// ---------------------------------------------------------------------------
// cat = [f1(64) ; msf(18) ; 0-pad to 96], point-major f16
// ---------------------------------------------------------------------------
__global__ void k_catpack(const _Float16* __restrict__ F1H,
                          const float* __restrict__ msf,
                          _Float16* __restrict__ CATH) {
  int t = blockIdx.x * blockDim.x + threadIdx.x;
  int n = t / 96, j = t % 96;
  _Float16 v = (_Float16)0.f;
  if (j < 64)       v = F1H[(size_t)n * 64 + j];
  else if (j < 82)  v = (_Float16)msf[(size_t)(j - 64) * NPTS + n];
  CATH[(size_t)n * 96 + j] = v;
}

// ---------------------------------------------------------------------------
// Point-wise WMMA GEMM: Y = act(W @ X + b).
// One wave = one 16-column tile, ALL row tiles (B registers reused nMt times).
// Packed W async-staged into LDS once per block (shared by 4 waves).
// X point-major [N][Kpad] f16, Y point-major [N][Mpad] f16 (+opt fp32 ch-major)
// ---------------------------------------------------------------------------
template <int Mpad, int Kpad, int M, int ACT>
__global__ void k_gemm_pw_t(const _Float16* __restrict__ Wp,
                            const float* __restrict__ bias,
                            const _Float16* __restrict__ Xh,
                            _Float16* __restrict__ Yh,
                            float* __restrict__ Yf32) {
  constexpr int nMt = Mpad / 16, nKt = Kpad / 32;
  __shared__ __align__(16) _Float16 wlds[nMt * nKt * 512];
  stage_lds(Wp, wlds, nMt * nKt * 1024, threadIdx.x, 128);
  __syncthreads();
  int w = threadIdx.x >> 5, L = threadIdx.x & 31;
  int ct = blockIdx.x * 4 + w;
  int col = (ct << 4) + (L & 15);
  int kbase = (L < 16) ? 0 : 16;
  v8f acc[nMt];
#pragma unroll
  for (int mt = 0; mt < nMt; ++mt) acc[mt] = (v8f){};
  for (int kt = 0; kt < nKt; ++kt) {
    v16h b = *(const v16h*)(Xh + (size_t)col * Kpad + kt * 32 + kbase);
#pragma unroll
    for (int mt = 0; mt < nMt; ++mt) {
      v16h a = *(const v16h*)(wlds + ((mt * nKt + kt) * 32 + L) * 16);
      acc[mt] = wmma32(a, b, acc[mt]);
    }
  }
#pragma unroll
  for (int mt = 0; mt < nMt; ++mt) {
    int rbase = (mt << 4) + ((L < 16) ? 0 : 8);
#pragma unroll
    for (int v = 0; v < 8; ++v) {
      int row = rbase + v;
      float val = 0.f;                       // padded rows -> clean zeros
      if (row < M) {
        val = acc[mt][v] + bias[row];
        if (ACT) val = fmaxf(val, 0.f);
      }
      Yh[(size_t)col * Mpad + row] = (_Float16)val;
      if (Yf32 && row < M) Yf32[(size_t)row * NPTS + col] = val;
    }
  }
}

// ---------------------------------------------------------------------------
// fsq[n] = sum_c feat[c][n]^2
// ---------------------------------------------------------------------------
__global__ void k_fsq(const float* __restrict__ FEAT, float* __restrict__ FSQ) {
  int n = blockIdx.x * blockDim.x + threadIdx.x;
  float s = 0.f;
  for (int c = 0; c < 20; ++c) {
    float v = FEAT[(size_t)c * NPTS + n];
    s += v * v;
  }
  FSQ[n] = s;
}

// c0 = feat[:, :256]  (CBUF is [32][256] fp32, rows 20..31 zero)
__global__ void k_c0(const float* __restrict__ FEAT, float* __restrict__ CBUF) {
  int t = blockIdx.x * blockDim.x + threadIdx.x;   // 8192
  int cc = t >> 8, s = t & 255;
  CBUF[cc * 256 + s] = (cc < 20) ? FEAT[(size_t)cc * NPTS + s] : 0.f;
}

__global__ void k_csq(const float* __restrict__ CBUF, float* __restrict__ CSQ) {
  int s = blockIdx.x * blockDim.x + threadIdx.x;   // 256
  float t = 0.f;
  for (int c = 0; c < 20; ++c) { float v = CBUF[c * 256 + s]; t += v * v; }
  CSQ[s] = t;
}

// ---------------------------------------------------------------------------
// G[s][n] = sum_c c[c][s] * feat[c][n]   (M=256, N=8192, K=32 pad of 20)
// One wave per column tile: single B load feeds 16 WMMAs (all s row-tiles).
// Packed c^T A-tiles (16KB) async-staged into LDS per block.
// ---------------------------------------------------------------------------
__global__ void k_d2dot(const _Float16* __restrict__ CPACKA,
                        const _Float16* __restrict__ FEATPM,
                        float* __restrict__ G) {
  __shared__ __align__(16) _Float16 alds[16 * 512];  // 16 A-tiles = 16KB
  stage_lds(CPACKA, alds, 16 * 1024, threadIdx.x, 128);
  __syncthreads();
  int w = threadIdx.x >> 5, L = threadIdx.x & 31;
  int ct = blockIdx.x * 4 + w;                 // 512 column tiles
  int col = (ct << 4) + (L & 15);
  int kbase = (L < 16) ? 0 : 16;
  v16h b = *(const v16h*)(FEATPM + (size_t)col * 32 + kbase);
#pragma unroll
  for (int mt = 0; mt < 16; ++mt) {
    v16h a = *(const v16h*)(alds + (mt * 32 + L) * 16);
    v8f acc = wmma32(a, b, (v8f){});
    int rb = (mt << 4) + ((L < 16) ? 0 : 8);
#pragma unroll
    for (int v = 0; v < 8; ++v)
      G[(size_t)(rb + v) * NPTS + col] = acc[v];
  }
}

// softmax over s of (2*G - csq - fsq); QH f16; optional fp32 Q to d_out
__global__ void k_softmax(const float* __restrict__ G,
                          const float* __restrict__ CSQ,
                          const float* __restrict__ FSQ,
                          _Float16* __restrict__ QH,
                          float* __restrict__ Qout, int writeQ) {
  int n = blockIdx.x * blockDim.x + threadIdx.x;
  float fs = FSQ[n];
  float m = -1e30f;
  for (int s = 0; s < NSP; ++s) {
    float v = 2.f * G[(size_t)s * NPTS + n] - CSQ[s] - fs;
    m = fmaxf(m, v);
  }
  float sum = 0.f;
  for (int s = 0; s < NSP; ++s) {
    float v = 2.f * G[(size_t)s * NPTS + n] - CSQ[s] - fs;
    sum += __expf(v - m);
  }
  float inv = 1.f / sum;
  for (int s = 0; s < NSP; ++s) {
    float v = 2.f * G[(size_t)s * NPTS + n] - CSQ[s] - fs;
    float q = __expf(v - m) * inv;
    QH[(size_t)s * NPTS + n] = (_Float16)q;
    if (writeQ) Qout[(size_t)s * NPTS + n] = q;
  }
}

// wave32 row reduction: sumQ[s] = sum_n QH[s][n]
__global__ void k_rowsum(const _Float16* __restrict__ QH,
                         float* __restrict__ SUMQ) {
  int w = threadIdx.x >> 5, L = threadIdx.x & 31;
  int s = blockIdx.x * 8 + w;
  float t = 0.f;
  for (int i = L; i < NPTS; i += 32) t += (float)QH[(size_t)s * NPTS + i];
  for (int off = 16; off > 0; off >>= 1) t += __shfl_down(t, off, 32);
  if (L == 0) SUMQ[s] = t;
}

// Cnum[c][s] = sum_n feat[c][n] * Q[s][n]  (K=8192 chained WMMA, B reused x2)
__global__ void k_gemm2(const _Float16* __restrict__ FPACKA,
                        const _Float16* __restrict__ QH,
                        float* __restrict__ CNUM) {
  int wave = blockIdx.x * 4 + (threadIdx.x >> 5);   // 16 column tiles
  int L = threadIdx.x & 31;
  int col = (wave << 4) + (L & 15);                 // superpixel s
  int kbase = (L < 16) ? 0 : 16;
  v8f acc[2] = {};
  for (int kt = 0; kt < NPTS / 32; ++kt) {
    if (kt + 8 < NPTS / 32)
      __builtin_prefetch(QH + (size_t)col * NPTS + (kt + 8) * 32, 0, 0);
    v16h b = *(const v16h*)(QH + (size_t)col * NPTS + kt * 32 + kbase);
#pragma unroll
    for (int mt = 0; mt < 2; ++mt) {
      v16h a = *(const v16h*)(FPACKA +
                              (((size_t)mt * (NPTS / 32) + kt) * 32 + L) * 16);
      acc[mt] = wmma32(a, b, acc[mt]);
    }
  }
#pragma unroll
  for (int mt = 0; mt < 2; ++mt) {
    int rb = (mt << 4) + ((L < 16) ? 0 : 8);
#pragma unroll
    for (int v = 0; v < 8; ++v)
      CNUM[(rb + v) * 256 + col] = acc[mt][v];
  }
}

__global__ void k_cupdate(const float* __restrict__ CNUM,
                          const float* __restrict__ SUMQ,
                          float* __restrict__ CBUF,
                          float* __restrict__ cout, int writeC) {
  int t = blockIdx.x * blockDim.x + threadIdx.x;   // 8192
  int cc = t >> 8, s = t & 255;
  float v = (cc < 20) ? CNUM[cc * 256 + s] / (SUMQ[s] + 1e-16f) : 0.f;
  CBUF[cc * 256 + s] = v;
  if (writeC && cc < 20) cout[cc * 256 + s] = v;
}

// ---------------------------------------------------------------------------
// Orchestration
// ---------------------------------------------------------------------------
extern "C" void kernel_launch(void* const* d_in, const int* in_sizes, int n_in,
                              void* d_out, int out_size, void* d_ws, size_t ws_size,
                              hipStream_t stream) {
  const float* x   = (const float*)d_in[0];
  const float* w1a = (const float*)d_in[1];
  const float* b1a = (const float*)d_in[2];
  const float* w1b = (const float*)d_in[3];
  const float* b1b = (const float*)d_in[4];
  const float* wm1 = (const float*)d_in[5];
  const float* bm1 = (const float*)d_in[6];
  const float* wm2 = (const float*)d_in[7];
  const float* bm2 = (const float*)d_in[8];
  const float* w2a = (const float*)d_in[9];
  const float* b2a = (const float*)d_in[10];
  const float* w2b = (const float*)d_in[11];
  const float* b2b = (const float*)d_in[12];
  const float* w3a = (const float*)d_in[13];
  const float* b3a = (const float*)d_in[14];
  const float* w3b = (const float*)d_in[15];
  const float* b3b = (const float*)d_in[16];
  const float* w3c = (const float*)d_in[17];
  const float* b3c = (const float*)d_in[18];

  char* ws = (char*)d_ws;
  // workspace layout (all offsets 32B+ aligned)
  _Float16* F1H    = (_Float16*)(ws + 0x0000000);   // 1 MB  [N][64]
  int*      IDXb   = (int*)     (ws + 0x0100000);   // 1 MB  [N][32]
  float*    D2N    = (float*)   (ws + 0x0200000);   // 1 MB  [N][32]
  _Float16* CATH   = (_Float16*)(ws + 0x0300000);   // 1.5MB [N][96]
  _Float16* F2AH   = (_Float16*)(ws + 0x0480000);   // 2 MB  [N][128]
  _Float16* F2H    = (_Float16*)(ws + 0x0680000);   // 2 MB  [N][128]
  _Float16* F3AH   = (_Float16*)(ws + 0x0880000);   // 2 MB  [N][128]
  _Float16* F3BH   = (_Float16*)(ws + 0x0A80000);   // 1 MB  [N][64]
  _Float16* FEATPM = (_Float16*)(ws + 0x0B80000);   // 512KB [N][32]
  float*    FEAT   = (float*)   (ws + 0x0C00000);   // 640KB [20][N]
  float*    FSQ    = (float*)   (ws + 0x0CA0000);   // 32KB
  float*    CSQ    = (float*)   (ws + 0x0CA8000);
  float*    SUMQ   = (float*)   (ws + 0x0CA8400);
  float*    CBUF   = (float*)   (ws + 0x0CA8800);   // 32KB [32][256]
  float*    CNUM   = (float*)   (ws + 0x0CB0800);   // 32KB [32][256]
  _Float16* WPwm1  = (_Float16*)(ws + 0x0CB8800);   // 8KB
  _Float16* WPw2a  = (_Float16*)(ws + 0x0CBA800);   // 24KB
  _Float16* WPw2b  = (_Float16*)(ws + 0x0CC0800);   // 32KB
  _Float16* WPw3a  = (_Float16*)(ws + 0x0CC8800);   // 32KB
  _Float16* WPw3b  = (_Float16*)(ws + 0x0CD0800);   // 16KB
  _Float16* WPw3c  = (_Float16*)(ws + 0x0CD4800);   // 4KB
  _Float16* CPACKA = (_Float16*)(ws + 0x0CD6000);   // 16KB (c^T A-tiles)
  _Float16* FPACKA = (_Float16*)(ws + 0x0CDA000);   // 512KB (feat A-tiles)
  _Float16* QH     = (_Float16*)(ws + 0x0D5A000);   // 4 MB  [256][N] f16
  float*    G      = (float*)   (ws + 0x115A000);   // 8 MB  [256][N] (also H2)
  float*    H2     = G;                             // reused: [262144][8]

  float* outQ = (float*)d_out;                      // [256][8192]
  float* outC = outQ + NSP * NPTS;                  // [20][256]
  float* outM = outC + 20 * NSP;                    // [18][8192]

  // 1. f1
  k_f1<<<NPTS / 256, 256, 0, stream>>>(x, w1a, b1a, w1b, b1b, F1H);
  // 2. knn
  k_knn<<<NPTS / 256, 256, 0, stream>>>(x, D2N, IDXb);
  // 3. pack weights into WMMA A-layout
  k_pack_w<<<(4096 + 255) / 256, 256, 0, stream>>>(wm1, WPwm1, 64, 64, 64, 64);
  k_pack_w<<<(12288 + 255) / 256, 256, 0, stream>>>(w2a, WPw2a, 128, 82, 128, 96);
  k_pack_w<<<(16384 + 255) / 256, 256, 0, stream>>>(w2b, WPw2b, 128, 128, 128, 128);
  k_pack_w<<<(16384 + 255) / 256, 256, 0, stream>>>(w3a, WPw3a, 128, 128, 128, 128);
  k_pack_w<<<(8192 + 255) / 256, 256, 0, stream>>>(w3b, WPw3b, 64, 128, 64, 128);
  k_pack_w<<<(2048 + 255) / 256, 256, 0, stream>>>(w3c, WPw3c, 20, 64, 32, 64);
  // 4. neighbor MLP (WMMA) -> H2, then multi-scale max -> msf (in d_out)
  k_edge<<<(NPTS * KNN / 16) / 4, 128, 0, stream>>>(WPwm1, bm1, wm2, bm2, F1H,
                                                    IDXb, H2);
  k_msf<<<NPTS / 256, 256, 0, stream>>>(H2, D2N, outM);
  // 5. cat = [f1; msf; 0]
  k_catpack<<<(NPTS * 96) / 256, 256, 0, stream>>>(F1H, outM, CATH);
  // 6. point-wise WMMA GEMM chain (128 blocks x 128 thr; 4 col-tiles/block)
  k_gemm_pw_t<128,  96, 128, 1><<<128, 128, 0, stream>>>(WPw2a, b2a, CATH, F2AH, nullptr);
  k_gemm_pw_t<128, 128, 128, 1><<<128, 128, 0, stream>>>(WPw2b, b2b, F2AH, F2H, nullptr);
  k_gemm_pw_t<128, 128, 128, 1><<<128, 128, 0, stream>>>(WPw3a, b3a, F2H, F3AH, nullptr);
  k_gemm_pw_t< 64, 128,  64, 1><<<128, 128, 0, stream>>>(WPw3b, b3b, F3AH, F3BH, nullptr);
  k_gemm_pw_t< 32,  64,  20, 0><<<128, 128, 0, stream>>>(WPw3c, b3c, F3BH, FEATPM, FEAT);
  // 7. slic prep: fsq, A-pack of feat (K=8192), c0
  k_fsq<<<NPTS / 256, 256, 0, stream>>>(FEAT, FSQ);
  k_pack_w<<<(2 * 256 * 512) / 256, 256, 0, stream>>>(FEAT, FPACKA, 20, NPTS, 32, NPTS);
  k_c0<<<8192 / 256, 256, 0, stream>>>(FEAT, CBUF);
  // 8. soft-SLIC: 10 updates + final assignment
  for (int it = 0; it <= 10; ++it) {
    k_csq<<<1, 256, 0, stream>>>(CBUF, CSQ);
    k_pack_cT<<<8192 / 256, 256, 0, stream>>>(CBUF, CPACKA);
    k_d2dot<<<512 / 4, 128, 0, stream>>>(CPACKA, FEATPM, G);
    k_softmax<<<NPTS / 256, 256, 0, stream>>>(G, CSQ, FSQ, QH, outQ,
                                              (it == 10) ? 1 : 0);
    if (it < 10) {
      k_rowsum<<<NSP / 8, 256, 0, stream>>>(QH, SUMQ);
      k_gemm2<<<16 / 4, 128, 0, stream>>>(FPACKA, QH, CNUM);
      k_cupdate<<<8192 / 256, 256, 0, stream>>>(CNUM, SUMQ, CBUF, outC,
                                                (it == 9) ? 1 : 0);
    }
  }
}
